// MARQuantizer2D_3779571220611
// MI455X (gfx1250) — matrix operand order repaired and checked by reading, not compiled, not run on hardware
//
#include <hip/hip_runtime.h>
#include <hip/hip_bf16.h>

// MI455X / gfx1250, wave32. Core compute = bf16 WMMA (v_wmma_f32_16x16x32_bf16):
//  - codebook distance GEMM + argmin (22.9 GFLOP, dominant), M-blocked 4x in
//    registers so each 16-code B tile feeds 4 independent WMMA chains
//    (L2 traffic /4, no acc RAW hazard between consecutive WMMAs)
//  - 3x3 conv as 9 shifted GEMMs over c_in (6 GFLOP)
// Codebook (4.2MB bf16) stays L2-resident (192MB); kernel is matrix-bound.

typedef __attribute__((ext_vector_type(16))) __bf16 v16bf;
typedef __attribute__((ext_vector_type(8)))  float  v8f;

#define DIM     128
#define NBCODE  16384
#define BATCH   16
#define NPIX    (BATCH*16*16)   // 4096
#define NELEM   (NPIX*DIM)      // 524288
#define NTOTIDX 5456

__device__ __forceinline__ unsigned short f2bf(float f) {
  unsigned u = __builtin_bit_cast(unsigned, f);
  u += 0x7FFFu + ((u >> 16) & 1u);           // round-to-nearest-even
  return (unsigned short)(u >> 16);
}

union V16U { uint4 u[2]; v16bf v; };

// A-operand (16-bit 16x32, CDNA5 ISA 7.12.2): lane<16 -> K {kb..kb+7, kb+16..kb+23},
// lane>=16 -> K {kb+8..kb+15, kb+24..kb+31}; row M = lane&15.
__device__ __forceinline__ v16bf load_a(const unsigned short* base, int row_base,
                                        int ld, int kb, int lane) {
  int m = lane & 15, hi = lane >> 4;
  const unsigned short* p = base + (size_t)(row_base + m) * ld + kb + hi * 8;
  V16U x;
  x.u[0] = *(const uint4*)(p);
  x.u[1] = *(const uint4*)(p + 16);
  return x.v;
}

// B-operand (16-bit 32x16): lanes 0-15 hold K {kb..kb+7, kb+16..kb+23}, lanes 16-31 K {kb+8..kb+15, kb+24..kb+31};
// column N = lane&15. Same K order as the A operand.
__device__ __forceinline__ v16bf load_b(const unsigned short* base, int col_base,
                                        int ld, int kb, int lane) {
  int n = lane & 15, hi = lane >> 4;
  const unsigned short* p = base + (size_t)(col_base + n) * ld + kb + hi * 8;
  V16U x;
  x.u[0] = *(const uint4*)(p);
  x.u[1] = *(const uint4*)(p + 16);
  return x.v;
}

// conv A with zero-padding on the l (width) border; keeps EXEC all-ones.
__device__ __forceinline__ v16bf load_a_conv(const unsigned short* base, int pix_row_base,
                                             int kb, int lane, int dl) {
  int m = lane & 15, hi = lane >> 4;
  int l2 = m + dl - 1;
  bool ok = (l2 >= 0) && (l2 < 16);
  int pix = pix_row_base + (ok ? l2 : 0);
  const unsigned short* p = base + (size_t)pix * DIM + kb + hi * 8;
  V16U x;
  x.u[0] = *(const uint4*)(p);
  x.u[1] = *(const uint4*)(p + 16);
  if (!ok) { x.u[0] = uint4{0,0,0,0}; x.u[1] = uint4{0,0,0,0}; }
  return x.v;
}

// ---------------- prep kernels ----------------

__global__ void k_prep_emb(const float* __restrict__ emb, unsigned short* __restrict__ ebf,
                           float* __restrict__ esq) {
  int code = blockIdx.x * 8 + (threadIdx.x >> 5);
  int lane = threadIdx.x & 31;
  const float* src = emb + (size_t)code * DIM;
  float ss = 0.f;
  for (int i = lane; i < DIM; i += 32) {
    float v = src[i];
    ebf[(size_t)code * DIM + i] = f2bf(v);
    ss += v * v;
  }
  for (int off = 16; off; off >>= 1) ss += __shfl_xor(ss, off, 32);
  if (lane == 0) esq[code] = ss;
}

// repack phi_w [4][co][ci][3][3] f32 -> [4][3*3][co][ci] bf16 (ci contiguous)
__global__ void k_prep_wpack(const float* __restrict__ w, unsigned short* __restrict__ wp) {
  int t = blockIdx.x * blockDim.x + threadIdx.x;
  if (t >= 4 * 9 * DIM * DIM) return;
  int ci = t & 127;
  int co = (t >> 7) & 127;
  int dd = (t >> 14) % 9;
  int k  = t / (DIM * DIM * 9);
  wp[((((size_t)k * 9 + dd) * DIM + co) * DIM) + ci] =
      f2bf(w[(((size_t)(k * DIM + co) * DIM + ci) * 9) + dd]);
}

// f_rest (NHWC) = f (NCHW); f_hat = 0
__global__ void k_init(const float* __restrict__ f, float* __restrict__ frest,
                       float* __restrict__ fhat) {
  int t = blockIdx.x * blockDim.x + threadIdx.x;
  if (t >= NELEM) return;
  int c = t & 127, l = (t >> 7) & 15, j = (t >> 11) & 15, b = t >> 15;
  frest[t] = f[(((b * DIM + c) * 16 + j) * 16 + l)];
  fhat[t]  = 0.f;
}

// ---------------- per-scale kernels ----------------

// area-downsample f_rest(NHWC) -> r [N,128] bf16 (N = 16*sn*sn, row order b,s,t)
__global__ void k_down(const float* __restrict__ frest, unsigned short* __restrict__ rbf, int sn) {
  int t = blockIdx.x * blockDim.x + threadIdx.x;
  int c = t & 127, n = t >> 7;
  int b = n / (sn * sn), rem = n % (sn * sn), s = rem / sn, tt = rem % sn;
  int blk = 16 / sn;
  float sum = 0.f;
  for (int jj = 0; jj < blk; ++jj)
    for (int ll = 0; ll < blk; ++ll)
      sum += frest[(((b * 16 + s * blk + jj) * 16 + tt * blk + ll) << 7) + c];
  rbf[(size_t)n * DIM + c] = f2bf(sum / (float)(blk * blk));
}

// WMMA codebook search, M-blocked by MT tiles of 16 rows per workgroup.
// 8 waves/WG, each wave scans its disjoint 2048-code slice against all MT A tiles.
// argmin(|r|^2 + e_sq - 2 r.e) == argmin(e_sq - 2 r.e)
template <int MT>
__global__ void k_argmin_t(const unsigned short* __restrict__ rbf,
                           const unsigned short* __restrict__ ebf,
                           const float* __restrict__ esq, int* __restrict__ idx_out) {
  __shared__ float svals[16 * MT * 128];
  __shared__ int   sidxs[16 * MT * 128];
  int lane = threadIdx.x & 31, wave = threadIdx.x >> 5;
  int row_base = blockIdx.x * (16 * MT);

  v16bf a[MT][4];
#pragma unroll
  for (int mt = 0; mt < MT; ++mt)
#pragma unroll
    for (int ks = 0; ks < 4; ++ks)
      a[mt][ks] = load_a(rbf, row_base + 16 * mt, DIM, ks * 32, lane);

  float best[MT][8];
  int   bidx[MT][8];
#pragma unroll
  for (int mt = 0; mt < MT; ++mt)
#pragma unroll
    for (int v = 0; v < 8; ++v) { best[mt][v] = 3.4e38f; bidx[mt][v] = 0; }

  int code0 = wave * (NBCODE / 8);
  for (int it = 0; it < (NBCODE / 8) / 16; ++it) {
    int cb = code0 + it * 16;
    v16bf b0 = load_b(ebf, cb, DIM, 0,  lane);
    v16bf b1 = load_b(ebf, cb, DIM, 32, lane);
    v16bf b2 = load_b(ebf, cb, DIM, 64, lane);
    v16bf b3 = load_b(ebf, cb, DIM, 96, lane);
    // prefetch next iteration's B rows (global_prefetch path)
    int ncb = (it + 1 < (NBCODE / 8) / 16) ? cb + 16 : code0;
    __builtin_prefetch(ebf + (size_t)(ncb + (lane & 15)) * DIM, 0, 3);

    int code = cb + (lane & 15);
    float es = esq[code];
#pragma unroll
    for (int mt = 0; mt < MT; ++mt) {
      v8f acc = {};
      acc = __builtin_amdgcn_wmma_f32_16x16x32_bf16(false, a[mt][0], false, b0, (short)0, acc, false, false);
      acc = __builtin_amdgcn_wmma_f32_16x16x32_bf16(false, a[mt][1], false, b1, (short)0, acc, false, false);
      acc = __builtin_amdgcn_wmma_f32_16x16x32_bf16(false, a[mt][2], false, b2, (short)0, acc, false, false);
      acc = __builtin_amdgcn_wmma_f32_16x16x32_bf16(false, a[mt][3], false, b3, (short)0, acc, false, false);
#pragma unroll
      for (int v = 0; v < 8; ++v) {
        float d = es - 2.0f * acc[v];
        if (d < best[mt][v]) { best[mt][v] = d; bidx[mt][v] = code; }
      }
    }
  }

  int col = wave * 16 + (lane & 15);
  int hi  = lane >> 4;
#pragma unroll
  for (int mt = 0; mt < MT; ++mt)
#pragma unroll
    for (int v = 0; v < 8; ++v) {
      int m = mt * 16 + v + 8 * hi;     // D layout: M = vgpr + 8*(lane/16), N = lane%16
      svals[m * 128 + col] = best[mt][v];
      sidxs[m * 128 + col] = bidx[mt][v];
    }
  __syncthreads();
  if (threadIdx.x < 16 * MT) {
    int m = threadIdx.x;
    float bv = svals[m * 128]; int bi = sidxs[m * 128];
    for (int c2 = 1; c2 < 128; ++c2) {
      float v = svals[m * 128 + c2]; int i2 = sidxs[m * 128 + c2];
      if (v < bv || (v == bv && i2 < bi)) { bv = v; bi = i2; }
    }
    idx_out[row_base + m] = bi;
  }
}

__device__ __forceinline__ float cubicw(float x) {
  const float a = -0.75f;
  float ax = fabsf(x);
  if (ax <= 1.f) return ((a + 2.f) * ax - (a + 3.f)) * ax * ax + 1.f;
  if (ax < 2.f)  return ((((ax - 5.f) * ax) + 8.f) * ax - 4.f) * a;
  return 0.f;
}

// gather emb[idx] and bicubic-upsample to 16x16; write h as f32 + bf16 (NHWC)
__global__ void k_upsample(const float* __restrict__ emb, const int* __restrict__ idxs,
                           float* __restrict__ hf, unsigned short* __restrict__ hbf, int sn) {
  int t = blockIdx.x * blockDim.x + threadIdx.x;
  if (t >= NELEM) return;
  int c = t & 127, pix = t >> 7;
  int l = pix & 15, j = (pix >> 4) & 15, b = pix >> 8;
  float val;
  if (sn == 16) {
    val = emb[(size_t)idxs[pix] * DIM + c];
  } else {
    float scale = (float)sn * (1.f / 16.f);
    float xj = (j + 0.5f) * scale - 0.5f; int x0j = (int)floorf(xj); float tj = xj - x0j;
    float xl = (l + 0.5f) * scale - 0.5f; int x0l = (int)floorf(xl); float tl = xl - x0l;
    val = 0.f;
    for (int ka = 0; ka < 4; ++ka) {
      int sj = min(max(x0j - 1 + ka, 0), sn - 1);
      float wj = cubicw(tj + 1.f - (float)ka);
      for (int kb = 0; kb < 4; ++kb) {
        int st = min(max(x0l - 1 + kb, 0), sn - 1);
        float wl = cubicw(tl + 1.f - (float)kb);
        val += wj * wl * emb[(size_t)idxs[(b * sn + sj) * sn + st] * DIM + c];
      }
    }
  }
  hf[t]  = val;
  hbf[t] = f2bf(val);
}

// WMMA 3x3 conv (9 shifted GEMMs over c_in) + residual mix + f_hat/f_rest update + SSE
__global__ void k_conv(const unsigned short* __restrict__ hbf, const float* __restrict__ hf,
                       const unsigned short* __restrict__ wp, const float* __restrict__ phib,
                       const float* __restrict__ f_in, float* __restrict__ fhat,
                       float* __restrict__ frest, float* __restrict__ sse_out, int kphi) {
  __shared__ float red[256];
  int lane = threadIdx.x & 31, wave = threadIdx.x >> 5;
  int tile = blockIdx.x;                 // 256 tiles of 16 pixels (one image row each)
  int b = tile >> 4, j = tile & 15;
  int cobase = wave * 16;                // 8 waves cover c_out = 128
  const unsigned short* wbase = wp + (size_t)kphi * 9 * DIM * DIM;

  v8f acc = {};
  for (int dj = 0; dj < 3; ++dj) {
    int j2 = j + dj - 1;
    if (j2 < 0 || j2 > 15) continue;     // uniform across wave: zero-pad row skipped
    int prb = (b * 16 + j2) * 16;
    for (int dl = 0; dl < 3; ++dl) {
      const unsigned short* wsl = wbase + (size_t)(dj * 3 + dl) * DIM * DIM;
#pragma unroll
      for (int ks = 0; ks < 4; ++ks) {
        v16bf a = load_a_conv(hbf, prb, ks * 32, lane, dl);
        v16bf bm = load_b(wsl, cobase, DIM, ks * 32, lane);
        acc = __builtin_amdgcn_wmma_f32_16x16x32_bf16(false, a, false, bm, (short)0, acc, false, false);
      }
    }
  }

  int hi = lane >> 4, n = lane & 15;
  int co = cobase + n;
  float bias = phib[kphi * DIM + co];
  float myerr = 0.f;
#pragma unroll
  for (int v = 0; v < 8; ++v) {
    int l = v + 8 * hi;                  // pixel (M) within tile
    int pix = (b * 16 + j) * 16 + l;
    size_t o = (size_t)pix * DIM + co;
    float y   = acc[v] + bias;
    float out = 0.5f * hf[o] + 0.5f * y;       // h*(1-RESI) + (conv+b)*RESI, RESI=0.5
    float nh  = fhat[o] + out;
    fhat[o]   = nh;
    frest[o] -= out;
    float e = nh - f_in[(((b * DIM + co) * 16 + j) * 16 + l)];
    myerr += e * e;
  }
  red[threadIdx.x] = myerr;
  __syncthreads();
  for (int s = 128; s; s >>= 1) {
    if (threadIdx.x < s) red[threadIdx.x] += red[threadIdx.x + s];
    __syncthreads();
  }
  if (threadIdx.x == 0) sse_out[blockIdx.x] = red[0];
}

// ---------------- epilogue kernels ----------------

__global__ void k_hist(const int* __restrict__ idx_all, unsigned* __restrict__ hist) {
  int t = blockIdx.x * blockDim.x + threadIdx.x;
  if (t < NTOTIDX) atomicAdd(&hist[idx_all[t]], 1u);
}

__global__ void k_out(const float* __restrict__ fhat, float* __restrict__ out) {
  int t = blockIdx.x * blockDim.x + threadIdx.x;  // NCHW flat
  if (t >= NELEM) return;
  int l = t & 15, j = (t >> 4) & 15, c = (t >> 8) & 127, b = t >> 15;
  out[t] = fhat[(((b * 16 + j) * 16 + l) << 7) + c];
}

__global__ void k_finalize(const unsigned* __restrict__ hist, const float* __restrict__ sse,
                           float* __restrict__ out) {
  __shared__ float r1[256], r2[256];
  float ent = 0.f;
  for (int i = threadIdx.x; i < NBCODE; i += 256) {
    float p = (float)hist[i] * (1.f / (float)NTOTIDX);
    ent += p * logf(p + 1e-10f);
  }
  float ls = 0.f;
  for (int i = threadIdx.x; i < 5 * 256; i += 256) ls += sse[i];
  r1[threadIdx.x] = ent; r2[threadIdx.x] = ls;
  __syncthreads();
  for (int s = 128; s; s >>= 1) {
    if (threadIdx.x < s) { r1[threadIdx.x] += r1[threadIdx.x + s]; r2[threadIdx.x] += r2[threadIdx.x + s]; }
    __syncthreads();
  }
  if (threadIdx.x == 0) {
    out[NELEM]     = r2[0] * (1.25f / ((float)NELEM * 5.f));  // (1+BETA)*mse summed / SN
    out[NELEM + 1] = expf(-r1[0]);
  }
}

// ---------------- launch ----------------

extern "C" void kernel_launch(void* const* d_in, const int* in_sizes, int n_in,
                              void* d_out, int out_size, void* d_ws, size_t ws_size,
                              hipStream_t stream) {
  (void)in_sizes; (void)n_in; (void)out_size; (void)ws_size;
  const float* f    = (const float*)d_in[0];   // [16,128,16,16]
  const float* emb  = (const float*)d_in[1];   // [16384,128]
  const float* phiw = (const float*)d_in[2];   // [4,128,128,3,3]
  const float* phib = (const float*)d_in[3];   // [4,128]
  float* out = (float*)d_out;

  char* ws = (char*)d_ws;
  size_t off = 0;
  auto alloc = [&](size_t bytes) -> void* {
    void* p = ws + off;
    off = (off + bytes + 255) & ~(size_t)255;
    return p;
  };
  unsigned short* ebf   = (unsigned short*)alloc((size_t)NBCODE * DIM * 2);
  unsigned short* wpack = (unsigned short*)alloc((size_t)4 * 9 * DIM * DIM * 2);
  float*          esq   = (float*)alloc((size_t)NBCODE * 4);
  float*          frest = (float*)alloc((size_t)NELEM * 4);
  float*          fhat  = (float*)alloc((size_t)NELEM * 4);
  float*          hf    = (float*)alloc((size_t)NELEM * 4);
  unsigned short* hbf   = (unsigned short*)alloc((size_t)NELEM * 2);
  unsigned short* rbf   = (unsigned short*)alloc((size_t)4096 * DIM * 2);
  int*            idxs  = (int*)alloc(8192 * 4);
  unsigned*       hist  = (unsigned*)alloc((size_t)NBCODE * 4);
  float*          sse   = (float*)alloc(5 * 256 * 4);

  k_prep_emb<<<NBCODE / 8, 256, 0, stream>>>(emb, ebf, esq);
  k_prep_wpack<<<(4 * 9 * DIM * DIM + 255) / 256, 256, 0, stream>>>(phiw, wpack);
  k_init<<<NELEM / 256, 256, 0, stream>>>(f, frest, fhat);

  const int SN[5]  = {1, 2, 4, 8, 16};
  const int OFF[5] = {0, 16, 80, 336, 1360};
  const int PHK[5] = {0, 1, 1, 2, 3};
  for (int si = 0; si < 5; ++si) {
    int sn = SN[si];
    int N  = 16 * sn * sn;
    k_down<<<(N * DIM) / 256, 256, 0, stream>>>(frest, rbf, sn);
    if (N >= 64) {
      k_argmin_t<4><<<N / 64, 256, 0, stream>>>(rbf, ebf, esq, idxs + OFF[si]);
    } else {
      k_argmin_t<1><<<N / 16, 256, 0, stream>>>(rbf, ebf, esq, idxs + OFF[si]);
    }
    k_upsample<<<NELEM / 256, 256, 0, stream>>>(emb, idxs + OFF[si], hf, hbf, sn);
    k_conv<<<256, 256, 0, stream>>>(hbf, hf, wpack, phib, f, fhat, frest,
                                    sse + si * 256, PHK[si]);
  }

  hipMemsetAsync(hist, 0, (size_t)NBCODE * 4, stream);
  k_hist<<<(NTOTIDX + 255) / 256, 256, 0, stream>>>(idxs, hist);
  k_out<<<NELEM / 256, 256, 0, stream>>>(fhat, out);
  k_finalize<<<1, 256, 0, stream>>>(hist, sse, out);
}
